// GPTModel_80152679678484
// MI455X (gfx1250) — compile-verified, matrix operand order checked by reading, not run on hardware
//
#include <hip/hip_runtime.h>
#include <hip/hip_bf16.h>

typedef __bf16 bf16_t;
typedef __bf16 v16bf __attribute__((ext_vector_type(16)));
typedef float  v8f   __attribute__((ext_vector_type(8)));

enum { Bc = 2, Sc = 1024, Dc = 1024, Hc = 16, HDc = 64, Lc = 6, Fc = 4096, Vc = 32000 };
enum { Mrows = Bc * Sc };           // 2048
enum { LDK = 40 };                  // LDS row pitch in bf16 (80 B, 16-byte aligned)

// Low 32 bits of a generic pointer to __shared__ data == LDS byte offset
// (ISA: LDS aperture decode uses addr[31:0] as the LDS address).
__device__ __forceinline__ unsigned lds_addr(const void* p)
{
    return (unsigned)(uintptr_t)p;
}

// ---------------------------------------------------------------------------
// Embedding + sinusoidal positional encoding:  x[b,s,d] = emb[id]*32 + pe(s,d)
// ---------------------------------------------------------------------------
__global__ __launch_bounds__(256)
void embed_pe(const int* __restrict__ ids, const float* __restrict__ emb,
              float* __restrict__ x)
{
    int bs  = blockIdx.x;               // 0 .. B*S-1
    int pos = bs & (Sc - 1);            // s = bs % S
    int tok = ids[bs];
    float fpos = (float)pos;
#pragma unroll
    for (int i = 0; i < Dc / 256; ++i) {
        int d = threadIdx.x + i * 256;
        float de  = (float)(d & ~1);
        float div = __expf(de * (-9.210340371976184f / (float)Dc)); // -ln(10000)/D
        float pe  = (d & 1) ? __cosf(fpos * div) : __sinf(fpos * div);
        x[(size_t)bs * Dc + d] = emb[(size_t)tok * Dc + d] * 32.0f + pe; // sqrt(1024)=32
    }
}

// ---------------------------------------------------------------------------
// LayerNorm, one block (256 threads) per row of D=1024
// ---------------------------------------------------------------------------
__global__ __launch_bounds__(256)
void layernorm(const float* __restrict__ x, const float* __restrict__ g,
               const float* __restrict__ b, float* __restrict__ out)
{
    __shared__ float red[16];
    int row = blockIdx.x;
    const float* xr = x + (size_t)row * Dc;
    float vals[4];
    float s1 = 0.f, s2 = 0.f;
#pragma unroll
    for (int i = 0; i < 4; ++i) {
        float v = xr[threadIdx.x + i * 256];
        vals[i] = v; s1 += v; s2 += v * v;
    }
#pragma unroll
    for (int off = 16; off; off >>= 1) {
        s1 += __shfl_xor(s1, off, 32);
        s2 += __shfl_xor(s2, off, 32);
    }
    int lane = threadIdx.x & 31, wid = threadIdx.x >> 5;
    if (lane == 0) { red[wid] = s1; red[8 + wid] = s2; }
    __syncthreads();
    float t1 = 0.f, t2 = 0.f;
#pragma unroll
    for (int i = 0; i < 8; ++i) { t1 += red[i]; t2 += red[8 + i]; }
    float mean = t1 * (1.0f / Dc);
    float var  = t2 * (1.0f / Dc) - mean * mean;
    float rstd = rsqrtf(var + 1e-5f);
#pragma unroll
    for (int i = 0; i < 4; ++i) {
        int d = threadIdx.x + i * 256;
        out[(size_t)row * Dc + d] = (vals[i] - mean) * rstd * g[d] + b[d];
    }
}

// ---------------------------------------------------------------------------
// WMMA bf16 fragment load from an LDS tile stored [row][k], pitch LDK=40.
// Per ISA: lanes 0-15 -> K pairs {0..7,16..23}; lanes 16-31 -> {8..15,24..31}.
// With an 80 B row pitch both 16 B K-groups are uint4-aligned: 2x ds_load_b128.
// Same per-lane indexing for A (row=M) and B-transposed (row=N).
// ---------------------------------------------------------------------------
__device__ __forceinline__ v16bf load_frag(const bf16_t* tile, int rowBase, int lane)
{
    int rr = rowBase + (lane & 15);
    int s  = lane >> 4;                                // K-half select
    const uint4* p = (const uint4*)(tile + rr * LDK);  // 16 B groups within row
    union { v16bf v; uint4 q[2]; } f;
    f.q[0] = p[s];           // K {0..7} or {8..15}
    f.q[1] = p[2 + s];       // K {16..23} or {24..31}
    return f.v;
}

// ---------------------------------------------------------------------------
// C[M,N] = act(A[M,K] @ W[K,N] + bias) (+ residual).  f32 in/out, bf16 WMMA.
// Block tile 128x128, K-step 32, 8 waves: wave = 32x64 (2x4 WMMA tiles).
// Double-buffered LDS; vectorized fills (float4 A loads, packed bf16 stores).
// ---------------------------------------------------------------------------
__global__ __launch_bounds__(256)
void gemm_bias_act(const float* __restrict__ A, const float* __restrict__ W,
                   const float* __restrict__ bias, const float* __restrict__ residual,
                   float* __restrict__ C, int M, int N, int K, int gelu_act, int nt_store)
{
    __shared__ bf16_t sA[2][128 * LDK];  // [m][k]
    __shared__ bf16_t sB[2][128 * LDK];  // [n][k] (transposed on fill)

    const int t    = threadIdx.x;
    const int lane = t & 31;
    const int wid  = t >> 5;
    const int blockM = blockIdx.y * 128;
    const int blockN = blockIdx.x * 128;
    const int wm = (wid >> 1) * 32;      // wave M offset in tile
    const int wn = (wid & 1)  * 64;      // wave N offset in tile

    float4 ra4[4];
    float  rb0[8], rb1[8];
    auto load_regs = [&](int k0) {
#pragma unroll
        for (int i = 0; i < 4; ++i) {                 // A: 1024 float4 quads
            int flat = t + i * 256;
            int r = flat >> 3, c4 = (flat & 7) * 4;
            ra4[i] = *(const float4*)&A[(size_t)(blockM + r) * K + k0 + c4];
        }
#pragma unroll
        for (int i = 0; i < 8; ++i) {                 // B: 2048 k-pairs
            int flat = t + i * 256;
            int kr2 = flat >> 7, cn = flat & 127;
            const float* wp = &W[(size_t)(k0 + kr2 * 2) * N + blockN + cn];
            rb0[i] = wp[0];
            rb1[i] = wp[N];
        }
    };
    auto store_lds = [&](int buf) {
#pragma unroll
        for (int i = 0; i < 4; ++i) {
            int flat = t + i * 256;
            int r = flat >> 3, c4 = (flat & 7) * 4;
            union { bf16_t h[4]; uint2 u; } pk;
            pk.h[0] = (bf16_t)ra4[i].x; pk.h[1] = (bf16_t)ra4[i].y;
            pk.h[2] = (bf16_t)ra4[i].z; pk.h[3] = (bf16_t)ra4[i].w;
            *(uint2*)&sA[buf][r * LDK + c4] = pk.u;   // one b64 LDS store
        }
#pragma unroll
        for (int i = 0; i < 8; ++i) {
            int flat = t + i * 256;
            int kr2 = flat >> 7, cn = flat & 127;
            union { bf16_t h[2]; unsigned u; } pk;
            pk.h[0] = (bf16_t)rb0[i]; pk.h[1] = (bf16_t)rb1[i];
            *(unsigned*)&sB[buf][cn * LDK + kr2 * 2] = pk.u;  // one b32 LDS store
        }
    };

    v8f acc[2][4];
#pragma unroll
    for (int i = 0; i < 2; ++i)
#pragma unroll
        for (int j = 0; j < 4; ++j) {
            v8f z = {0.f, 0.f, 0.f, 0.f, 0.f, 0.f, 0.f, 0.f};
            acc[i][j] = z;
        }

    load_regs(0);
    store_lds(0);
    __syncthreads();

    const int T = K >> 5;
    int buf = 0;
    for (int tix = 0; tix < T; ++tix) {
        bool more = (tix + 1) < T;
        if (more) load_regs((tix + 1) << 5);      // global loads in flight...

        v16bf af[2], bfr[4];                      // ...while WMMAs run
#pragma unroll
        for (int tm = 0; tm < 2; ++tm) af[tm]  = load_frag(sA[buf], wm + tm * 16, lane);
#pragma unroll
        for (int tn = 0; tn < 4; ++tn) bfr[tn] = load_frag(sB[buf], wn + tn * 16, lane);
#pragma unroll
        for (int tm = 0; tm < 2; ++tm)
#pragma unroll
            for (int tn = 0; tn < 4; ++tn)
                acc[tm][tn] = __builtin_amdgcn_wmma_f32_16x16x32_bf16(
                    false, af[tm], false, bfr[tn], (short)0, acc[tm][tn], false, false);

        if (more) store_lds(buf ^ 1);
        __syncthreads();
        buf ^= 1;
    }

    // Epilogue: C-layout lane L -> N = L&15, M = r + 8*(L>>4) for VGPR r.
    const int coln  = lane & 15;
    const int rbase = (lane >> 4) * 8;
#pragma unroll
    for (int tm = 0; tm < 2; ++tm) {
#pragma unroll
        for (int tn = 0; tn < 4; ++tn) {
            int n = blockN + wn + tn * 16 + coln;
            float bia = bias[n];
#pragma unroll
            for (int r = 0; r < 8; ++r) {
                int m = blockM + wm + tm * 16 + rbase + r;
                float v = acc[tm][tn][r] + bia;
                if (gelu_act) v = 0.5f * v * (1.0f + erff(v * 0.70710678118f));
                size_t off = (size_t)m * N + n;
                if (residual) v += residual[off];
                if (nt_store) __builtin_nontemporal_store(v, &C[off]);
                else          C[off] = v;
            }
        }
    }
}

// ---------------------------------------------------------------------------
// Causal attention, online softmax. Block = 8 waves = 8 query rows of one
// (b,h). K/V tiles (64 keys x 64 dims, f32) DMA'd to LDS with
// GLOBAL_LOAD_ASYNC_TO_LDS_B128 (ASYNCcnt), lane owns 2 of the 64 head dims.
// ---------------------------------------------------------------------------
__global__ __launch_bounds__(256)
void attention(const float* __restrict__ q, const float* __restrict__ k,
               const float* __restrict__ v, float* __restrict__ ctx)
{
    __shared__ float sK[64][64];
    __shared__ float sV[64][64];
    int bh = blockIdx.y;                 // 0 .. B*H-1
    int b  = bh >> 4;
    int h  = bh & 15;
    int wid = threadIdx.x >> 5, lane = threadIdx.x & 31;
    int qrow = blockIdx.x * 8 + wid;
    int d0 = lane * 2;
    size_t qoff = ((size_t)(b * Sc + qrow)) * Dc + h * HDc + d0;
    float q0 = q[qoff]     * 0.125f;     // 1/sqrt(64)
    float q1 = q[qoff + 1] * 0.125f;
    float mx = -3.0e38f, l = 0.f, a0 = 0.f, a1 = 0.f;
    int qmaxblk = blockIdx.x * 8 + 7;

    for (int kt = 0; kt <= qmaxblk; kt += 64) {
        __syncthreads();                 // previous tile fully consumed
        // Async DMA the 16 KB K tile + 16 KB V tile straight into LDS:
        // 4 x b128 per thread per tile (1024 b128 transfers across 256 threads).
#pragma unroll
        for (int i = 0; i < 4; ++i) {
            int flat = threadIdx.x + i * 256;            // 0..1023 b128 units
            int j  = flat >> 4;                          // key row in tile
            int c4 = (flat & 15) * 4;                    // dim group of 4 floats
            const float* gk = k + ((size_t)(b * Sc + kt + j)) * Dc + h * HDc + c4;
            const float* gv = v + ((size_t)(b * Sc + kt + j)) * Dc + h * HDc + c4;
            unsigned lk = lds_addr(&sK[j][c4]);
            unsigned lv = lds_addr(&sV[j][c4]);
            asm volatile("global_load_async_to_lds_b128 %0, %1, off"
                         :: "v"(lk), "v"(gk) : "memory");
            asm volatile("global_load_async_to_lds_b128 %0, %1, off"
                         :: "v"(lv), "v"(gv) : "memory");
        }
#if __has_builtin(__builtin_amdgcn_s_wait_asynccnt)
        __builtin_amdgcn_s_wait_asynccnt(0);
#else
        asm volatile("s_wait_asynccnt 0x0" ::: "memory");
#endif
        __syncthreads();                 // all waves' DMAs complete & visible

        int jmax = qrow - kt;            // inclusive causal bound within tile
        if (jmax > 63) jmax = 63;
        for (int j = 0; j <= jmax; ++j) {
            float p = q0 * sK[j][d0] + q1 * sK[j][d0 + 1];
#pragma unroll
            for (int off = 16; off; off >>= 1) p += __shfl_xor(p, off, 32);
            float nm = fmaxf(mx, p);
            float sc = __expf(mx - nm);
            float e  = __expf(p - nm);
            l  = l  * sc + e;
            a0 = a0 * sc + e * sV[j][d0];
            a1 = a1 * sc + e * sV[j][d0 + 1];
            mx = nm;
        }
    }
    float inv = 1.0f / l;
    ctx[qoff]     = a0 * inv;
    ctx[qoff + 1] = a1 * inv;
}

// ---------------------------------------------------------------------------
extern "C" void kernel_launch(void* const* d_in, const int* in_sizes, int n_in,
                              void* d_out, int out_size, void* d_ws, size_t ws_size,
                              hipStream_t stream)
{
    (void)in_sizes; (void)n_in; (void)out_size; (void)ws_size;
    const int*   ids   = (const int*)  d_in[0];
    const float* emb   = (const float*)d_in[1];
    const float* Wq    = (const float*)d_in[2];
    const float* bq    = (const float*)d_in[3];
    const float* Wk    = (const float*)d_in[4];
    const float* bk    = (const float*)d_in[5];
    const float* Wv    = (const float*)d_in[6];
    const float* bv    = (const float*)d_in[7];
    const float* Wo    = (const float*)d_in[8];
    const float* bo    = (const float*)d_in[9];
    const float* ln1g  = (const float*)d_in[10];
    const float* ln1b  = (const float*)d_in[11];
    const float* ln2g  = (const float*)d_in[12];
    const float* ln2b  = (const float*)d_in[13];
    const float* W1    = (const float*)d_in[14];
    const float* b1    = (const float*)d_in[15];
    const float* W2    = (const float*)d_in[16];
    const float* b2    = (const float*)d_in[17];
    const float* fng   = (const float*)d_in[18];
    const float* fnb   = (const float*)d_in[19];
    const float* Wout  = (const float*)d_in[20];
    const float* bout  = (const float*)d_in[21];

    float* ws = (float*)d_ws;
    const size_t OFF = (size_t)Mrows * Dc;      // 2,097,152 floats
    float* x  = ws;
    float* hb = ws + OFF;
    float* qb = ws + 2 * OFF;
    float* kb = ws + 3 * OFF;
    float* vb = ws + 4 * OFF;
    float* cb = ws + 5 * OFF;
    float* fb = ws + 6 * OFF;                   // needs M*F = 4*OFF floats

    embed_pe<<<Mrows, 256, 0, stream>>>(ids, emb, x);

    dim3 gD(Dc / 128, Mrows / 128);             // (8, 16)
    dim3 gF(Fc / 128, Mrows / 128);             // (32, 16)
    dim3 gV(Vc / 128, Mrows / 128);             // (250, 16)
    dim3 gA(Sc / 8, Bc * Hc);                   // (128, 32)

    for (int i = 0; i < Lc; ++i) {
        const float* wq = Wq + (size_t)i * Dc * Dc;
        const float* wk = Wk + (size_t)i * Dc * Dc;
        const float* wv = Wv + (size_t)i * Dc * Dc;
        const float* wo = Wo + (size_t)i * Dc * Dc;
        const float* w1 = W1 + (size_t)i * Dc * Fc;
        const float* w2 = W2 + (size_t)i * Fc * Dc;

        layernorm<<<Mrows, 256, 0, stream>>>(x, ln1g + (size_t)i * Dc, ln1b + (size_t)i * Dc, hb);
        gemm_bias_act<<<gD, 256, 0, stream>>>(hb, wq, bq + (size_t)i * Dc, nullptr, qb, Mrows, Dc, Dc, 0, 0);
        gemm_bias_act<<<gD, 256, 0, stream>>>(hb, wk, bk + (size_t)i * Dc, nullptr, kb, Mrows, Dc, Dc, 0, 0);
        gemm_bias_act<<<gD, 256, 0, stream>>>(hb, wv, bv + (size_t)i * Dc, nullptr, vb, Mrows, Dc, Dc, 0, 0);
        attention<<<gA, 256, 0, stream>>>(qb, kb, vb, cb);
        gemm_bias_act<<<gD, 256, 0, stream>>>(cb, wo, bo + (size_t)i * Dc, x, x, Mrows, Dc, Dc, 0, 0);
        layernorm<<<Mrows, 256, 0, stream>>>(x, ln2g + (size_t)i * Dc, ln2b + (size_t)i * Dc, hb);
        gemm_bias_act<<<gF, 256, 0, stream>>>(hb, w1, b1 + (size_t)i * Fc, nullptr, fb, Mrows, Fc, Dc, 1, 0);
        gemm_bias_act<<<gD, 256, 0, stream>>>(fb, w2, b2 + (size_t)i * Dc, x, x, Mrows, Dc, Fc, 0, 0);
    }

    layernorm<<<Mrows, 256, 0, stream>>>(x, fng, fnb, hb);
    gemm_bias_act<<<gV, 256, 0, stream>>>(hb, Wout, bout, nullptr, (float*)d_out,
                                          Mrows, Vc, Dc, 0, 1);
}
